// SNNModel_2413771620717
// MI455X (gfx1250) — matrix operand order repaired and checked by reading, not compiled, not run on hardware
//
#include <hip/hip_runtime.h>
#include <hip/hip_bf16.h>

typedef __attribute__((ext_vector_type(16))) _Float16 v16h_t;
typedef __attribute__((ext_vector_type(8)))  _Float16 v8h_t;
typedef __attribute__((ext_vector_type(8)))  float    v8f_t;
typedef __attribute__((ext_vector_type(4)))  int      v4i_t;

// Address-space-qualified element types for the async-LDS builtin
typedef __attribute__((address_space(1))) v4i_t as1_v4i;
typedef __attribute__((address_space(3))) v4i_t as3_v4i;

// Problem constants (match the reference)
#define BB   4096
#define DD   1024
#define LL1  2048
#define LL2  3072
#define OO   128
#define TT   15
#define BN_EPS 1e-5f

// LDS tile: 128 rows x 32 K-halves, padded to 72 halves/row (144B) ->
// conflict-free, 16B-aligned b128 fragment reads.
#define LDS_STRIDE 72

// ---------------------------------------------------------------------------
// CDNA5 async global->LDS copy (ASYNCcnt-tracked), with portable fallback.
// ---------------------------------------------------------------------------
#if __has_builtin(__builtin_amdgcn_global_load_async_to_lds_b128)
#define HAVE_ASYNC_LDS 1
#else
#define HAVE_ASYNC_LDS 0
#endif

__device__ __forceinline__ void copy16_to_lds(_Float16* lds_dst,
                                              const _Float16* gsrc) {
#if HAVE_ASYNC_LDS
    __builtin_amdgcn_global_load_async_to_lds_b128(
        (as1_v4i*)(gsrc),          // global source (addrspace(1) int4*)
        (as3_v4i*)(lds_dst),       // LDS destination (addrspace(3) int4*)
        /*offset=*/0, /*cpol=*/0);
#else
    *(v8h_t*)lds_dst = *(const v8h_t*)gsrc;
#endif
}

__device__ __forceinline__ void async_wait0() {
#if HAVE_ASYNC_LDS
#if __has_builtin(__builtin_amdgcn_s_wait_asynccnt)
    __builtin_amdgcn_s_wait_asynccnt(0);
#else
    asm volatile("s_wait_asynccnt 0" ::: "memory");
#endif
#endif
}

// ---------------------------------------------------------------------------
// f32 -> f16 conversion
// ---------------------------------------------------------------------------
__global__ __launch_bounds__(256) void cvt_f32_to_f16(const float* __restrict__ in,
                                                      _Float16* __restrict__ out,
                                                      long n) {
    long i = (long)blockIdx.x * 256 + threadIdx.x;
    if (i < n) out[i] = (_Float16)in[i];
}

// ---------------------------------------------------------------------------
// WMMA fragment loaders (from LDS)
// A-matrix 16x32 f16 layout (ISA 7.12.2): lane<16 -> M=lane, K=0..7,16..23 ;
// lane>=16 -> M=lane-16, K=8..15,24..31
// ---------------------------------------------------------------------------
__device__ __forceinline__ v16h_t frag_a_lds(const _Float16* tile16, int lane) {
    const int r  = lane & 15;
    const int kb = (lane >> 4) * 8;
    const _Float16* p = tile16 + r * LDS_STRIDE + kb;
    v8h_t lo = *(const v8h_t*)(p);        // K = kb .. kb+7
    v8h_t hi = *(const v8h_t*)(p + 16);   // K = kb+16 .. kb+23
    v16h_t o;
#pragma unroll
    for (int i = 0; i < 8; ++i) { o[i] = lo[i]; o[i + 8] = hi[i]; }
    return o;
}

// B-matrix 32x16 f16: lane<16 -> N=lane, K=0..7,16..23 ; lane>=16 -> N=lane-16, K=8..15,24..31
// (tile stored as [N][K], i.e. weight rows)
__device__ __forceinline__ v16h_t frag_b_lds(const _Float16* tile16, int lane) {
    const int n  = lane & 15;
    const int kb = (lane >> 4) * 8;
    const _Float16* p = tile16 + n * LDS_STRIDE + kb;
    v8h_t lo = *(const v8h_t*)(p);
    v8h_t hi = *(const v8h_t*)(p + 16);
    v16h_t o;
#pragma unroll
    for (int i = 0; i < 8; ++i) { o[i] = lo[i]; o[i + 8] = hi[i]; }
    return o;
}

// ---------------------------------------------------------------------------
// Tiled f16 WMMA GEMM:  Z[M,N] = A[M,K] @ Bm[N,K]^T   (Bm row-major = weights)
// Workgroup: 256 thr (8 waves), 128x128 tile; wave = 64x32 patch = 4x2 WMMAs.
// Double-buffered LDS; next tile staged via async global->LDS while WMMAs run
// on the current buffer. FUSE3: fused layer-3 leaky (reset='none') epilogue.
// ---------------------------------------------------------------------------
template <bool FUSE3>
__global__ __launch_bounds__(256) void gemm_wmma_f16(
    const _Float16* __restrict__ A, const _Float16* __restrict__ Bm,
    float* __restrict__ Z, int M, int N, int K,
    const float* __restrict__ bias, float* __restrict__ mstate,
    const float* __restrict__ beta_p, const float* __restrict__ thr_p,
    float* __restrict__ spk, float* __restrict__ mem) {

    __shared__ _Float16 lA[2][128 * LDS_STRIDE];
    __shared__ _Float16 lB[2][128 * LDS_STRIDE];

    const int tid  = threadIdx.x;
    const int lane = tid & 31;
    const int wave = tid >> 5;
    const int wm   = (wave & 1) * 64;   // wave M offset in tile
    const int wn   = (wave >> 1) * 32;  // wave N offset in tile
    const int m0   = blockIdx.y * 128;
    const int n0   = blockIdx.x * 128;
    const int nk   = K >> 5;

    // Stage one 128x32 A tile + 128x32 B tile into LDS buffer `buf`.
    auto stage = [&](int buf, int k0) {
#pragma unroll
        for (int i = 0; i < 2; ++i) {
            int c   = tid + i * 256;       // 0..511 chunk id
            int row = c >> 2;
            int col = (c & 3) * 8;         // halves
            copy16_to_lds(&lA[buf][row * LDS_STRIDE + col],
                          &A[(size_t)(m0 + row) * K + k0 + col]);
            copy16_to_lds(&lB[buf][row * LDS_STRIDE + col],
                          &Bm[(size_t)(n0 + row) * K + k0 + col]);
        }
    };

    stage(0, 0);
    async_wait0();
    __syncthreads();

    v8f_t acc[4][2] = {};

    for (int kt = 0; kt < nk; ++kt) {
        const int cur = kt & 1;
        if (kt + 1 < nk) stage(cur ^ 1, (kt + 1) * 32);   // overlap with compute

        const _Float16* la = &lA[cur][0];
        const _Float16* lb = &lB[cur][0];
        v16h_t aF[4], bF[2];
#pragma unroll
        for (int mi = 0; mi < 4; ++mi)
            aF[mi] = frag_a_lds(la + (wm + mi * 16) * LDS_STRIDE, lane);
#pragma unroll
        for (int ni = 0; ni < 2; ++ni)
            bF[ni] = frag_b_lds(lb + (wn + ni * 16) * LDS_STRIDE, lane);

#pragma unroll
        for (int mi = 0; mi < 4; ++mi)
#pragma unroll
            for (int ni = 0; ni < 2; ++ni)
                acc[mi][ni] = __builtin_amdgcn_wmma_f32_16x16x32_f16(
                    false, aF[mi], false, bF[ni], (short)0, acc[mi][ni],
                    false, false);

        async_wait0();     // our async copies for the next buffer are in LDS
        __syncthreads();   // everyone done reading cur + sees next buffer
    }

    // Epilogue. C/D layout: lane<16 -> N=lane, M=r ; lane>=16 -> N=lane-16, M=8+r
    const int rbase = (lane >> 4) * 8;
    const int nl    = lane & 15;
    const int gm0   = m0 + wm + rbase;
    const int gn0   = n0 + wn + nl;

    if (!FUSE3) {
        float* p0 = Z + (size_t)gm0 * N + gn0;
#pragma unroll
        for (int mi = 0; mi < 4; ++mi)
#pragma unroll
            for (int r = 0; r < 8; ++r) {
                float* pr = p0 + (mi * 16 + r) * N;   // 32-bit relative offset
#pragma unroll
                for (int ni = 0; ni < 2; ++ni)
                    pr[ni * 16] = acc[mi][ni][r];
            }
    } else {
        const float bta = fminf(fmaxf(*beta_p, 0.f), 1.f);
        const float thr = *thr_p;
        const size_t base = (size_t)gm0 * N + gn0;
#pragma unroll
        for (int ni = 0; ni < 2; ++ni) {
            const float bsv = bias[gn0 + ni * 16];
#pragma unroll
            for (int mi = 0; mi < 4; ++mi)
#pragma unroll
                for (int r = 0; r < 8; ++r) {
                    const size_t idx = base + (unsigned)((mi * 16 + r) * N + ni * 16);
                    float mn = bta * mstate[idx] + (acc[mi][ni][r] + bsv);
                    mstate[idx] = mn;                               // reset='none'
                    spk[idx] = (mn - thr) > 0.f ? 1.f : 0.f;
                    mem[idx] = mn;
                }
        }
    }
}

// ---------------------------------------------------------------------------
// BatchNorm stats over batch axis: column mean + rsqrt(var+eps).
// Block: 256 thr = 64 cols x 4 row-lanes.
// ---------------------------------------------------------------------------
__global__ __launch_bounds__(256) void bn_stats(const float* __restrict__ Z,
                                                int Mrows, int N,
                                                float* __restrict__ mu,
                                                float* __restrict__ rstd) {
    __shared__ float ssum[256];
    __shared__ float ssq[256];
    const int tx  = threadIdx.x & 63;
    const int ty  = threadIdx.x >> 6;
    const int col = blockIdx.x * 64 + tx;
    float s = 0.f, q = 0.f;
    for (int r = ty; r < Mrows; r += 4) {
        float v = Z[(size_t)r * N + col];
        s += v; q += v * v;
    }
    ssum[threadIdx.x] = s;
    ssq[threadIdx.x]  = q;
    __syncthreads();
    if (ty == 0) {
        s = ssum[tx] + ssum[tx + 64] + ssum[tx + 128] + ssum[tx + 192];
        q = ssq[tx]  + ssq[tx + 64]  + ssq[tx + 128]  + ssq[tx + 192];
        float mean = s / (float)Mrows;
        float var  = q / (float)Mrows - mean * mean;   // biased, as torch BN
        mu[col]   = mean;
        rstd[col] = rsqrtf(var + BN_EPS);
    }
}

// ---------------------------------------------------------------------------
// Fused BN-apply + leaky(subtract reset) + spike; writes f16 spikes for the
// next WMMA GEMM. (b1/b2 cancel inside BN, so Z has no bias — exact math.)
// ---------------------------------------------------------------------------
__global__ __launch_bounds__(256) void leaky_bn(
    const float* __restrict__ Z, const float* __restrict__ mu,
    const float* __restrict__ rstd, const float* __restrict__ g,
    const float* __restrict__ bbeta, float* __restrict__ mstate,
    _Float16* __restrict__ sOut, const float* __restrict__ beta_p,
    const float* __restrict__ thr_p, int N, long total) {
    long idx = (long)blockIdx.x * 256 + threadIdx.x;
    if (idx >= total) return;
    const int j = (int)(idx % N);
    const float bta = fminf(fmaxf(*beta_p, 0.f), 1.f);
    const float thr = *thr_p;
    float cur = g[j] * (Z[idx] - mu[j]) * rstd[j] + bbeta[j];
    float mo  = mstate[idx];
    float rst = (mo - thr) > 0.f ? thr : 0.f;       // reset from OLD mem
    float mn  = bta * mo + cur - rst;
    mstate[idx] = mn;
    sOut[idx] = (_Float16)((mn - thr) > 0.f ? 1.f : 0.f);
}

// ---------------------------------------------------------------------------
extern "C" void kernel_launch(void* const* d_in, const int* in_sizes, int n_in,
                              void* d_out, int out_size, void* d_ws, size_t ws_size,
                              hipStream_t stream) {
    (void)in_sizes; (void)n_in; (void)out_size; (void)ws_size;
    const float* x   = (const float*)d_in[0];
    const float* W1  = (const float*)d_in[1];
    /* b1 = d_in[2] — cancels in BN */
    const float* W2  = (const float*)d_in[3];
    /* b2 = d_in[4] — cancels in BN */
    const float* W3  = (const float*)d_in[5];
    const float* b3  = (const float*)d_in[6];
    const float* g1  = (const float*)d_in[7];
    const float* bb1 = (const float*)d_in[8];
    const float* g2  = (const float*)d_in[9];
    const float* bb2 = (const float*)d_in[10];
    const float* beta1 = (const float*)d_in[11];
    const float* thr1  = (const float*)d_in[12];
    const float* beta2 = (const float*)d_in[13];
    const float* thr2  = (const float*)d_in[14];
    const float* beta3 = (const float*)d_in[15];
    const float* thr3  = (const float*)d_in[16];

    // ---- workspace carve (256B aligned) ----
    char* w = (char*)d_ws;
    auto carve = [&](size_t bytes) {
        char* p = w;
        w += (bytes + 255) & ~(size_t)255;
        return p;
    };
    _Float16* xh  = (_Float16*)carve((size_t)BB * DD * 2);
    _Float16* W1h = (_Float16*)carve((size_t)LL1 * DD * 2);
    _Float16* W2h = (_Float16*)carve((size_t)LL2 * LL1 * 2);
    _Float16* W3h = (_Float16*)carve((size_t)OO * LL2 * 2);
    float*    z1  = (float*)   carve((size_t)BB * LL1 * 4);
    float*    z2  = (float*)   carve((size_t)BB * LL2 * 4);
    _Float16* s1  = (_Float16*)carve((size_t)BB * LL1 * 2);
    _Float16* s2  = (_Float16*)carve((size_t)BB * LL2 * 2);
    float*    m1  = (float*)   carve((size_t)BB * LL1 * 4);
    float*    m2  = (float*)   carve((size_t)BB * LL2 * 4);
    float*    m3  = (float*)   carve((size_t)BB * OO * 4);
    float*    mu1   = (float*)carve((size_t)LL1 * 4);
    float*    rstd1 = (float*)carve((size_t)LL1 * 4);
    float*    mu2   = (float*)carve((size_t)LL2 * 4);
    float*    rstd2 = (float*)carve((size_t)LL2 * 4);

    float* spk = (float*)d_out;                    // [T, B, O]
    float* mem = spk + (size_t)TT * BB * OO;       // [T, B, O]

    // ---- zero membrane states (capturable) ----
    (void)hipMemsetAsync(m1, 0, (size_t)BB * LL1 * 4, stream);
    (void)hipMemsetAsync(m2, 0, (size_t)BB * LL2 * 4, stream);
    (void)hipMemsetAsync(m3, 0, (size_t)BB * OO * 4, stream);

    // ---- f32 -> f16 casts (spikes are exactly representable later) ----
    auto cvt = [&](const float* in, _Float16* out, long n) {
        cvt_f32_to_f16<<<(unsigned)((n + 255) / 256), 256, 0, stream>>>(in, out, n);
    };
    cvt(x,  xh,  (long)BB * DD);
    cvt(W1, W1h, (long)LL1 * DD);
    cvt(W2, W2h, (long)LL2 * LL1);
    cvt(W3, W3h, (long)OO * LL2);

    // ---- time-invariant layer-1 pre-activation + BN stats (once) ----
    gemm_wmma_f16<false><<<dim3(LL1 / 128, BB / 128), 256, 0, stream>>>(
        xh, W1h, z1, BB, LL1, DD, nullptr, nullptr, nullptr, nullptr, nullptr, nullptr);
    bn_stats<<<LL1 / 64, 256, 0, stream>>>(z1, BB, LL1, mu1, rstd1);

    // ---- temporal loop ----
    for (int t = 0; t < TT; ++t) {
        leaky_bn<<<(unsigned)(((long)BB * LL1 + 255) / 256), 256, 0, stream>>>(
            z1, mu1, rstd1, g1, bb1, m1, s1, beta1, thr1, LL1, (long)BB * LL1);

        gemm_wmma_f16<false><<<dim3(LL2 / 128, BB / 128), 256, 0, stream>>>(
            s1, W2h, z2, BB, LL2, LL1, nullptr, nullptr, nullptr, nullptr, nullptr, nullptr);

        bn_stats<<<LL2 / 64, 256, 0, stream>>>(z2, BB, LL2, mu2, rstd2);

        leaky_bn<<<(unsigned)(((long)BB * LL2 + 255) / 256), 256, 0, stream>>>(
            z2, mu2, rstd2, g2, bb2, m2, s2, beta2, thr2, LL2, (long)BB * LL2);

        gemm_wmma_f16<true><<<dim3(OO / 128, BB / 128), 256, 0, stream>>>(
            s2, W3h, nullptr, BB, OO, LL2, b3, m3, beta3, thr3,
            spk + (size_t)t * BB * OO, mem + (size_t)t * BB * OO);
    }
}
